// BertSelfAttention_69630009803421
// MI455X (gfx1250) — matrix operand order repaired and checked by reading, not compiled, not run on hardware
//
#include <hip/hip_runtime.h>
#include <math.h>
#include <stdint.h>

// ---------------------------------------------------------------------------
// BERT self-attention for MI455X (gfx1250, wave32, WMMA + async LDS DMA).
// B=4, S=2048, NH=16, DH=64, HID=1024.
// All matmuls via V_WMMA_F32_16X16X32_BF16.  Score gemm fuses the segment
// bias via a 128-wide contraction: QA row = [q*scale | q+b_q_s],
// KB row = [k | seg].  Shared tiles (weights, K, V) are staged to LDS with
// GLOBAL_LOAD_ASYNC_TO_LDS_B128 (ASYNCcnt), double-buffered.
// ---------------------------------------------------------------------------

#define NHD   16
#define DHD   64
#define HIDD  1024
#define BAT   4
#define SEQ   2048
#define NROW  (BAT * SEQ)   // 8192
#define KDIM  128           // fused contraction for the score gemm

typedef __attribute__((ext_vector_type(16))) __bf16 v16bf;
typedef __attribute__((ext_vector_type(8)))  __bf16 v8bf;
typedef __attribute__((ext_vector_type(8)))  float  v8f;

static __device__ __forceinline__ v8f bwmma(v16bf a, v16bf b, v8f c) {
  // 8 args: (neg_a, A, neg_b, B, c_mod, C, reuse_a, reuse_b)
  return __builtin_amdgcn_wmma_f32_16x16x32_bf16(false, a, false, b,
                                                 (short)0, c, false, false);
}

static __device__ __forceinline__ v16bf pack16(v8bf lo, v8bf hi) {
  return __builtin_shufflevector(lo, hi, 0, 1, 2, 3, 4, 5, 6, 7,
                                 8, 9, 10, 11, 12, 13, 14, 15);
}

// Async DMA of 16 bytes global -> LDS (VGLOBAL GLOBAL_LOAD_ASYNC_TO_LDS_B128,
// tracked by ASYNCcnt).  ldsPtr is a pointer into a __shared__ array; the LDS
// byte address is its flat address truncated to 32 bits (ISA aperture rule).
static __device__ __forceinline__ void async_b128(const void* gptr,
                                                  void* ldsPtr) {
  uint32_t lofs = (uint32_t)(uintptr_t)ldsPtr;
  asm volatile("global_load_async_to_lds_b128 %0, %1, off"
               :: "v"(lofs), "v"((uint64_t)(uintptr_t)gptr)
               : "memory");
}

static __device__ __forceinline__ void wait_async0() {
  asm volatile("s_wait_asynccnt 0" ::: "memory");
}

// ---------------------------------------------------------------------------
// Kernel 1: f32 -> bf16 conversion (activations + weights)
// ---------------------------------------------------------------------------
__global__ void cvt_f32_bf16(const float* __restrict__ in,
                             __bf16* __restrict__ out, int n) {
  int i = blockIdx.x * blockDim.x + threadIdx.x;
  if (i < n) out[i] = (__bf16)in[i];
}

// ---------------------------------------------------------------------------
// Kernel 2: segment-embedding gather into KB[..., 64:128]
// ---------------------------------------------------------------------------
__global__ void seg_fill(const int* __restrict__ seg_ids,
                         const float* __restrict__ seg_table,
                         __bf16* __restrict__ KB) {
  int t = blockIdx.x * blockDim.x + threadIdx.x;   // over B*S*NH
  if (t >= BAT * SEQ * NHD) return;
  int h = t % NHD;
  int bs = t / NHD;
  int s = bs % SEQ, b = bs / SEQ;
  int id = seg_ids[b * SEQ + s];
  const float* src = seg_table + (size_t)id * HIDD + h * DHD;
  __bf16* dst = KB + ((size_t)(b * NHD + h) * SEQ + s) * KDIM + DHD;
#pragma unroll 8
  for (int d = 0; d < DHD; ++d) dst[d] = (__bf16)src[d];
}

// ---------------------------------------------------------------------------
// Kernel 3: QKV projection gemm.  out[m,n] = sum_k hb[m,k] * W[n,k]
// Block = 256 threads (8 waves); tile M=128, N=64.  Weight tile (64K x 64N,
// 8 KiB) staged to LDS via async DMA, double-buffered; A rows loaded directly
// (unique per wave).  Epilogue scatters to QA / KB / VT with biases + scale.
// ---------------------------------------------------------------------------
__global__ __launch_bounds__(256, 1) void qkv_gemm(
    const __bf16* __restrict__ hb, const __bf16* __restrict__ wb,
    const float* __restrict__ bq, const float* __restrict__ bv,
    const float* __restrict__ bqs,
    __bf16* __restrict__ QA, __bf16* __restrict__ KB, __bf16* __restrict__ VT) {
  __shared__ __attribute__((aligned(128))) __bf16 Bt[2][64 * 64];  // 16 KiB

  const int tid  = threadIdx.x;
  const int lane = tid & 31, wave = tid >> 5;
  const int half = lane >> 4, l16 = lane & 15;
  const int mrow = blockIdx.x * 128 + wave * 16;
  const int ngl  = blockIdx.y * 64;          // 0..3071
  const int widx = ngl / HIDD;               // 0=q, 1=k, 2=v
  const int nbase = ngl % HIDD;
  const __bf16* W = wb + (size_t)widx * HIDD * HIDD;

  // Stage one 64K x 64N weight tile ([n_local][k_local] row-major).
  auto stage = [&](int buf, int k0) {
#pragma unroll
    for (int i = 0; i < 2; ++i) {
      int e   = tid + i * 256;      // 512 x 16B chunks
      int n_l = e >> 3;             // 0..63
      int k8  = (e & 7) * 8;        // half offset within row
      async_b128(W + (size_t)(nbase + n_l) * HIDD + k0 + k8,
                 &Bt[buf][n_l * 64 + k8]);
    }
  };

  v8f acc[4] = {v8f{}, v8f{}, v8f{}, v8f{}};
  stage(0, 0);

  for (int k0 = 0; k0 < HIDD; k0 += 64) {
    const int buf = (k0 >> 6) & 1;
    wait_async0();
    __syncthreads();                       // tile[buf] fully in LDS
    if (k0 + 64 < HIDD) stage(buf ^ 1, k0 + 64);

#pragma unroll
    for (int sub = 0; sub < 64; sub += 32) {
      const __bf16* ap = hb + (size_t)(mrow + l16) * HIDD + k0 + sub + half * 16;
      v16bf A = pack16(*(const v8bf*)ap, *(const v8bf*)(ap + 8));
#pragma unroll
      for (int t = 0; t < 4; ++t) {
        const __bf16* bp = &Bt[buf][(t * 16 + l16) * 64 + sub + half * 16];
        v16bf Bm = *(const v16bf*)bp;
        acc[t] = bwmma(A, Bm, acc[t]);
      }
    }
    __syncthreads();                       // all reads done before overwrite
  }

  const float scale = 0.125f;  // 1/sqrt(64)
#pragma unroll
  for (int t = 0; t < 4; ++t) {
    int n = nbase + t * 16 + l16;
    int h = n >> 6, d = n & 63;
#pragma unroll
    for (int v = 0; v < 8; ++v) {
      int m = mrow + v + half * 8;     // C/D layout: rows v + 8*(lane/16)
      int b = m >> 11, s = m & 2047;
      size_t row = (size_t)(b * NHD + h) * SEQ + s;
      float val = acc[t][v];
      if (widx == 0) {
        val += bq[n];
        QA[row * KDIM + d]       = (__bf16)(val * scale);   // q*scale
        QA[row * KDIM + DHD + d] = (__bf16)(val + bqs[n]);  // q + b_q_s
      } else if (widx == 1) {
        KB[row * KDIM + d] = (__bf16)val;                   // k (no bias)
      } else {
        val += bv[n];
        VT[((size_t)(b * NHD + h) * DHD + d) * SEQ + s] = (__bf16)val;
      }
    }
  }
}

// ---------------------------------------------------------------------------
// Kernel 4: flash attention.  grid = (B*NH, SEQ/128), block = 256 (8 waves).
// Per iteration (32 keys): async-stage K tile (32x128) + V^T tile (64x32)
// to LDS (shared by all 8 waves, double-buffered), 8 score WMMAs (K=128),
// online softmax with shfl_xor row reductions, P relayout through per-wave
// LDS, 4 P@V WMMAs.
// ---------------------------------------------------------------------------
__global__ __launch_bounds__(256, 1) void flash_attn(
    const __bf16* __restrict__ QA, const __bf16* __restrict__ KB,
    const __bf16* __restrict__ VT, const float* __restrict__ mask,
    float* __restrict__ out) {
  __shared__ __attribute__((aligned(128))) __bf16 Kt[2][32 * 128]; // 16 KiB
  __shared__ __attribute__((aligned(128))) __bf16 Vt[2][64 * 32];  //  8 KiB
  __shared__ __attribute__((aligned(32)))  __bf16 plds[8][16][32]; //  8 KiB

  const int tid  = threadIdx.x;
  const int lane = tid & 31, wave = tid >> 5;
  const int half = lane >> 4, l16 = lane & 15;
  const int bh = blockIdx.x;        // b*NH + h
  const int b = bh >> 4, h = bh & 15;
  const int qbase = blockIdx.y * 128 + wave * 16;

  const float* mk = mask + (size_t)b * SEQ;
  const __bf16* kbase = KB + (size_t)bh * SEQ * KDIM;
  const __bf16* vbase = VT + (size_t)bh * DHD * SEQ;

  // Stage K tile (32 rows x 128 halves) + V^T tile (64 rows x 32 halves).
  auto stage = [&](int buf, int j) {
#pragma unroll
    for (int i = 0; i < 2; ++i) {
      int e  = tid + i * 256;     // 512 x 16B chunks
      int r  = e >> 4;            // key row 0..31
      int c8 = (e & 15) * 8;      // half offset within 128-row
      async_b128(kbase + (size_t)(j + r) * KDIM + c8, &Kt[buf][r * 128 + c8]);
    }
    {
      int r  = tid >> 2;          // d row 0..63
      int c8 = (tid & 3) * 8;     // key offset within 32-row
      async_b128(vbase + (size_t)r * SEQ + j + c8, &Vt[buf][r * 32 + c8]);
    }
  };

  // Load the wave's 16x128 QA strip once (4 K-chunks of 32).
  v16bf Aq[4];
  {
    const __bf16* ap = QA + ((size_t)bh * SEQ + qbase + l16) * KDIM;
#pragma unroll
    for (int c = 0; c < 4; ++c) {
      const __bf16* p = ap + c * 32 + half * 16;
      Aq[c] = pack16(*(const v8bf*)p, *(const v8bf*)(p + 8));
    }
  }

  v8f O[4] = {v8f{}, v8f{}, v8f{}, v8f{}};   // 16x64 f32 accumulator
  float mrow[8], lrow[8];
#pragma unroll
  for (int v = 0; v < 8; ++v) { mrow[v] = -__builtin_inff(); lrow[v] = 0.f; }

  stage(0, 0);

  for (int j = 0; j < SEQ; j += 32) {
    const int buf = (j >> 5) & 1;
    wait_async0();
    __syncthreads();                         // tiles[buf] fully in LDS
    if (j + 32 < SEQ) stage(buf ^ 1, j + 32);

    v8f S0 = v8f{}, S1 = v8f{};
#pragma unroll
    for (int c = 0; c < 4; ++c) {
      const __bf16* b0 = &Kt[buf][l16 * 128 + c * 32 + half * 16];
      const __bf16* b1 = b0 + 16 * 128;
      S0 = bwmma(Aq[c], *(const v16bf*)b0, S0);
      S1 = bwmma(Aq[c], *(const v16bf*)b1, S1);
    }
    const float am0 = mk[j + l16];
    const float am1 = mk[j + 16 + l16];

    // Online softmax: per-row (v) reductions across the 16 lanes of the half.
#pragma unroll
    for (int v = 0; v < 8; ++v) {
      S0[v] += am0;
      S1[v] += am1;
      float r = fmaxf(S0[v], S1[v]);
      r = fmaxf(r, __shfl_xor(r, 1, 32));
      r = fmaxf(r, __shfl_xor(r, 2, 32));
      r = fmaxf(r, __shfl_xor(r, 4, 32));
      r = fmaxf(r, __shfl_xor(r, 8, 32));
      float mnew = fmaxf(mrow[v], r);
      float alpha = __expf(mrow[v] - mnew);
      mrow[v] = mnew;
      float p0 = __expf(S0[v] - mnew);
      float p1 = __expf(S1[v] - mnew);
      S0[v] = p0; S1[v] = p1;
      float sm = p0 + p1;
      sm += __shfl_xor(sm, 1, 32);
      sm += __shfl_xor(sm, 2, 32);
      sm += __shfl_xor(sm, 4, 32);
      sm += __shfl_xor(sm, 8, 32);
      lrow[v] = lrow[v] * alpha + sm;
#pragma unroll
      for (int t = 0; t < 4; ++t) O[t][v] *= alpha;
    }

    // P (two 16x16 D tiles) -> per-wave LDS in [row][key] order.
#pragma unroll
    for (int v = 0; v < 8; ++v) {
      plds[wave][v + half * 8][l16]      = (__bf16)S0[v];
      plds[wave][v + half * 8][16 + l16] = (__bf16)S1[v];
    }
    asm volatile("s_wait_dscnt 0" ::: "memory");  // same-wave LDS RAW

    // Re-read as A operand 16x32 (keys = contraction dim).
    const __bf16* pp = &plds[wave][l16][half * 16];
    v16bf P = pack16(*(const v8bf*)pp, *(const v8bf*)(pp + 8));

#pragma unroll
    for (int t = 0; t < 4; ++t) {
      const __bf16* vp = &Vt[buf][(t * 16 + l16) * 32 + half * 16];
      O[t] = bwmma(P, *(const v16bf*)vp, O[t]);
    }
    __syncthreads();                         // reads done before overwrite
  }

  // Normalize and write ctx[b, s, h*64 + d] (f32).
#pragma unroll
  for (int v = 0; v < 8; ++v) {
    float inv = 1.0f / lrow[v];
    int srow = qbase + v + half * 8;
    float* op = out + ((size_t)b * SEQ + srow) * HIDD + h * DHD;
#pragma unroll
    for (int t = 0; t < 4; ++t) op[t * 16 + l16] = O[t][v] * inv;
  }
}

// ---------------------------------------------------------------------------
// Launch.  Workspace layout (bytes):
//   hb  @ 0        : 16 MiB  (bf16 hidden, 8192x1024)
//   wb  @ 16 MiB   :  6 MiB  (bf16 Wq|Wk|Wv, each 1024x1024)
//   QA  @ 22 MiB   : 32 MiB  (bf16 [b,h,s,128] = [q*scale | q+b_q_s])
//   KB  @ 54 MiB   : 32 MiB  (bf16 [b,h,s,128] = [k | seg])
//   VT  @ 86 MiB   : 16 MiB  (bf16 [b,h,d,s])
// ---------------------------------------------------------------------------
extern "C" void kernel_launch(void* const* d_in, const int* in_sizes, int n_in,
                              void* d_out, int out_size, void* d_ws,
                              size_t ws_size, hipStream_t stream) {
  const float* hidden = (const float*)d_in[0];
  const float* amask  = (const float*)d_in[1];
  const int*   segid  = (const int*)d_in[2];
  const float* Wq     = (const float*)d_in[3];
  const float* bq     = (const float*)d_in[4];
  const float* Wk     = (const float*)d_in[5];
  const float* Wv     = (const float*)d_in[6];
  const float* bv     = (const float*)d_in[7];
  const float* segtab = (const float*)d_in[8];
  const float* bqs    = (const float*)d_in[9];
  float* out = (float*)d_out;

  char* ws = (char*)d_ws;
  __bf16* hb = (__bf16*)(ws);
  __bf16* wb = (__bf16*)(ws + (16ull << 20));
  __bf16* QA = (__bf16*)(ws + (22ull << 20));
  __bf16* KB = (__bf16*)(ws + (54ull << 20));
  __bf16* VT = (__bf16*)(ws + (86ull << 20));

  const int nh = NROW * HIDD;      // 8M activations
  const int nw = HIDD * HIDD;      // 1M per weight
  cvt_f32_bf16<<<(nh + 255) / 256, 256, 0, stream>>>(hidden, hb, nh);
  cvt_f32_bf16<<<(nw + 255) / 256, 256, 0, stream>>>(Wq, wb, nw);
  cvt_f32_bf16<<<(nw + 255) / 256, 256, 0, stream>>>(Wk, wb + (size_t)nw, nw);
  cvt_f32_bf16<<<(nw + 255) / 256, 256, 0, stream>>>(Wv, wb + (size_t)2 * nw, nw);

  seg_fill<<<(BAT * SEQ * NHD + 255) / 256, 256, 0, stream>>>(segid, segtab, KB);

  dim3 g2(NROW / 128, (3 * HIDD) / 64);   // 64 x 48 blocks
  qkv_gemm<<<g2, 256, 0, stream>>>(hb, wb, bq, bv, bqs, QA, KB, VT);

  dim3 g3(BAT * NHD, SEQ / 128);          // 64 x 16 blocks
  flash_attn<<<g3, 256, 0, stream>>>(QA, KB, VT, amask, out);
}
